// MoIETransformerBlock_19679540150486
// MI455X (gfx1250) — matrix operand order repaired and checked by reading, not compiled, not run on hardware
//
#include <hip/hip_runtime.h>
#include <hip/hip_bf16.h>
#include <math.h>

// ---------------------------------------------------------------------------
// MoIE transformer block for MI455X (gfx1250, wave32, WMMA f32_16x16x32_f16)
//
// All contractions run on v_wmma_f32_16x16x32_f16 (f16 in / f32 acc) — the
// workload is compute-bound (~215 GFLOP vs ~20us of HBM traffic at 23.3TB/s),
// so matrix-core throughput is what matters. Pointwise math stays f32.
// ---------------------------------------------------------------------------

typedef __attribute__((ext_vector_type(16))) _Float16 v16h;
typedef __attribute__((ext_vector_type(8)))  _Float16 h8;
typedef __attribute__((ext_vector_type(4)))  _Float16 h4;
typedef __attribute__((ext_vector_type(8)))  float    v8f;

#define D_DIM   1024
#define S_DIM   2048
#define B_DIM   4
#define M_TOK   (B_DIM * S_DIM)   // 8192 tokens
#define SCALE_R 0.03125f          // 1/sqrt(1024)
#define LN_EPS  1e-5f
#define GATE_EPS 1e-9f

// GEMM tile: block = 256 threads = 8 waves, tile 128x64, K-step 32.
// 8 waves in a 4(m) x 2(n) grid; each wave owns 32x32 = 2x2 WMMA fragments.
#define BM 128
#define BN 64
#define BK 32
#define LDS_STRIDE 40   // halves; 80B rows keep all fragment loads 16B aligned

union Frag { v16h v; h8 h[2]; };

// ---------------------------------------------------------------------------
// Core: C[m0+..BM, n0+..BN] = A(MxK, row-major) . B(NxK, row-major)^T
// BTRANS=true : B supplied as KxN row-major (ldb = N row stride) -> staged
//               transposed into LDS (used for attn @ v).
// DUAL=true   : two B matrices contracted against the same A tile.
// Epi(row, col, acc0, acc1) performs the store + pointwise epilogue.
// ---------------------------------------------------------------------------
template <bool DUAL, bool BTRANS, class Epi>
__device__ __forceinline__ void gemm_nt(const float* __restrict__ A, int lda,
                                        const float* __restrict__ B0,
                                        const float* __restrict__ B1, int ldb,
                                        int K, int m0, int n0, Epi epi) {
  __shared__ _Float16 As[BM * LDS_STRIDE];
  __shared__ _Float16 Bs0[BN * LDS_STRIDE];
  __shared__ _Float16 Bs1[DUAL ? BN * LDS_STRIDE : 1];

  const int tid  = threadIdx.x;
  const int lane = tid & 31;
  const int wave = tid >> 5;
  const int wm   = wave >> 1;  // 0..3 : 32-row band
  const int wn   = wave & 1;   // 0..1 : 32-col band

  v8f acc0[2][2] = {};
  v8f acc1[2][2] = {};

  for (int k0 = 0; k0 < K; k0 += BK) {
    // -------- stage A tile (BM x BK), f32 -> f16 --------
    {
      const int kq = (tid & 7) * 4;
      const int rb = tid >> 3;
      if (k0 + BK < K)  // hint next K-slab into cache (global_prefetch_b8)
        __builtin_prefetch(A + (size_t)(m0 + rb) * lda + k0 + BK);
#pragma unroll
      for (int p = 0; p < 4; ++p) {
        const int r = rb + p * 32;
        const float4 f = *(const float4*)(A + (size_t)(m0 + r) * lda + k0 + kq);
        h4 hv;
        hv.x = (_Float16)f.x; hv.y = (_Float16)f.y;
        hv.z = (_Float16)f.z; hv.w = (_Float16)f.w;
        *(h4*)(As + r * LDS_STRIDE + kq) = hv;
      }
    }
    // -------- stage B tile(s) (BN x BK) --------
    if constexpr (!BTRANS) {
      const int kq = (tid & 7) * 4;
      const int rb = tid >> 3;
#pragma unroll
      for (int p = 0; p < 2; ++p) {
        const int n = rb + p * 32;
        const float4 f0 = *(const float4*)(B0 + (size_t)(n0 + n) * ldb + k0 + kq);
        h4 hv;
        hv.x = (_Float16)f0.x; hv.y = (_Float16)f0.y;
        hv.z = (_Float16)f0.z; hv.w = (_Float16)f0.w;
        *(h4*)(Bs0 + n * LDS_STRIDE + kq) = hv;
        if constexpr (DUAL) {
          const float4 f1 = *(const float4*)(B1 + (size_t)(n0 + n) * ldb + k0 + kq);
          h4 hw;
          hw.x = (_Float16)f1.x; hw.y = (_Float16)f1.y;
          hw.z = (_Float16)f1.z; hw.w = (_Float16)f1.w;
          *(h4*)(Bs1 + n * LDS_STRIDE + kq) = hw;
        }
      }
    } else {
      // B is KxN row-major; transpose while staging so contraction is
      // contiguous in LDS. Loads stay fully coalesced along N.
      const int n4 = (tid & 15) * 4;
      const int kb = tid >> 4;  // 0..15
#pragma unroll
      for (int p = 0; p < 2; ++p) {
        const int kk = kb + p * 16;
        const float4 f = *(const float4*)(B0 + (size_t)(k0 + kk) * ldb + n0 + n4);
        Bs0[(n4 + 0) * LDS_STRIDE + kk] = (_Float16)f.x;
        Bs0[(n4 + 1) * LDS_STRIDE + kk] = (_Float16)f.y;
        Bs0[(n4 + 2) * LDS_STRIDE + kk] = (_Float16)f.z;
        Bs0[(n4 + 3) * LDS_STRIDE + kk] = (_Float16)f.w;
      }
    }
    __syncthreads();

    // -------- fragments (contiguous K: lane half h holds k = 16h + s) --------
    Frag fa[2], fb0[2], fb1[2];
    {
      const int kc = lane >> 4;   // which K half this lane holds
      const int rr = lane & 15;
#pragma unroll
      for (int t = 0; t < 2; ++t) {
        const int row = wm * 32 + t * 16 + rr;
        // A: lane m row m; K halves {kc*16 .. +16}
        fa[t].h[0] = *(const h8*)(As + row * LDS_STRIDE + kc * 16);
        fa[t].h[1] = *(const h8*)(As + row * LDS_STRIDE + kc * 16 + 8);
        const int col = wn * 32 + t * 16 + rr;
        // B: lane n column n; K halves {kc*16 .. +16}
        fb0[t].h[0] = *(const h8*)(Bs0 + col * LDS_STRIDE + kc * 16);
        fb0[t].h[1] = *(const h8*)(Bs0 + col * LDS_STRIDE + kc * 16 + 8);
        if constexpr (DUAL) {
          fb1[t].h[0] = *(const h8*)(Bs1 + col * LDS_STRIDE + kc * 16);
          fb1[t].h[1] = *(const h8*)(Bs1 + col * LDS_STRIDE + kc * 16 + 8);
        }
      }
    }
#pragma unroll
    for (int tm = 0; tm < 2; ++tm)
#pragma unroll
      for (int tn = 0; tn < 2; ++tn) {
        acc0[tm][tn] = __builtin_amdgcn_wmma_f32_16x16x32_f16(
            false, fa[tm].v, false, fb0[tn].v, (short)0, acc0[tm][tn], false, false);
        if constexpr (DUAL)
          acc1[tm][tn] = __builtin_amdgcn_wmma_f32_16x16x32_f16(
              false, fa[tm].v, false, fb1[tn].v, (short)0, acc1[tm][tn], false, false);
      }
    __syncthreads();
  }

  // -------- epilogue: C layout VGPR j -> row j (lanes 0-15) / j+8 (16-31) --
  {
    const int rr  = lane & 15;
    const int rhi = (lane >> 4) * 8;
#pragma unroll
    for (int tm = 0; tm < 2; ++tm)
#pragma unroll
      for (int tn = 0; tn < 2; ++tn)
#pragma unroll
        for (int j = 0; j < 8; ++j) {
          const int gr = m0 + wm * 32 + tm * 16 + j + rhi;
          const int gc = n0 + wn * 32 + tn * 16 + rr;
          epi(gr, gc, acc0[tm][tn][j], acc1[tm][tn][j]);
        }
  }
}

// ---------------------------------------------------------------------------
// Block-wide reductions (256 threads)
// ---------------------------------------------------------------------------
__device__ __forceinline__ float block_sum(float v, float* sb) {
  const int tid = threadIdx.x;
  __syncthreads();
  sb[tid] = v;
  __syncthreads();
  for (int s = 128; s > 0; s >>= 1) {
    if (tid < s) sb[tid] += sb[tid + s];
    __syncthreads();
  }
  return sb[0];
}

__device__ __forceinline__ float block_max(float v, float* sb) {
  const int tid = threadIdx.x;
  __syncthreads();
  sb[tid] = v;
  __syncthreads();
  for (int s = 128; s > 0; s >>= 1) {
    if (tid < s) sb[tid] = fmaxf(sb[tid], sb[tid + s]);
    __syncthreads();
  }
  return sb[0];
}

__device__ __forceinline__ float silu_f(float z) {
  return z / (1.0f + __expf(-z));
}

// ---------------------------------------------------------------------------
// Pointwise kernels
// ---------------------------------------------------------------------------
__global__ void __launch_bounds__(256) k_cost(const float* __restrict__ gate,
                                              float* __restrict__ cost) {
  __shared__ float sb[256];
  const int n = blockIdx.x;
  const float4 v = *(const float4*)(gate + n * D_DIM + threadIdx.x * 4);
  float m = fmaxf(fmaxf(fabsf(v.x), fabsf(v.y)), fmaxf(fabsf(v.z), fabsf(v.w)));
  m = block_max(m, sb);
  const float inv = 1.0f / (m + GATE_EPS);
  float4 o;
  o.x = v.x * inv; o.y = v.y * inv; o.z = v.z * inv; o.w = v.w * inv;
  *(float4*)(cost + n * D_DIM + threadIdx.x * 4) = o;
}

__global__ void __launch_bounds__(256) k_ln_x(const float* __restrict__ x,
                                              const float* __restrict__ g,
                                              const float* __restrict__ b,
                                              float* __restrict__ h) {
  __shared__ float sb[256];
  const size_t row = blockIdx.x;
  const int c = threadIdx.x * 4;
  const float4 v = *(const float4*)(x + row * D_DIM + c);
  float s = v.x + v.y + v.z + v.w;
  float q = v.x * v.x + v.y * v.y + v.z * v.z + v.w * v.w;
  s = block_sum(s, sb);
  q = block_sum(q, sb);
  const float mean = s * (1.0f / D_DIM);
  const float var  = q * (1.0f / D_DIM) - mean * mean;
  const float inv  = rsqrtf(var + LN_EPS);
  const float4 gg = *(const float4*)(g + c);
  const float4 bb = *(const float4*)(b + c);
  float4 o;
  o.x = (v.x - mean) * inv * gg.x + bb.x;
  o.y = (v.y - mean) * inv * gg.y + bb.y;
  o.z = (v.z - mean) * inv * gg.z + bb.z;
  o.w = (v.w - mean) * inv * gg.w + bb.w;
  *(float4*)(h + row * D_DIM + c) = o;
}

// proto_state = proto_w + LN(prc)  (row-wise LN, done in place on prc)
__global__ void __launch_bounds__(256) k_proto_ln(float* __restrict__ prc,
                                                  const float* __restrict__ proto_w,
                                                  const float* __restrict__ pln_g,
                                                  const float* __restrict__ pln_b) {
  __shared__ float sb[256];
  const size_t row = blockIdx.x;        // n*D + e
  const int n = (int)(row >> 10);
  const int c = threadIdx.x * 4;
  const float4 v = *(const float4*)(prc + row * D_DIM + c);
  float s = v.x + v.y + v.z + v.w;
  float q = v.x * v.x + v.y * v.y + v.z * v.z + v.w * v.w;
  s = block_sum(s, sb);
  q = block_sum(q, sb);
  const float mean = s * (1.0f / D_DIM);
  const float var  = q * (1.0f / D_DIM) - mean * mean;
  const float inv  = rsqrtf(var + LN_EPS);
  const float4 gg = *(const float4*)(pln_g + n * D_DIM + c);
  const float4 bb = *(const float4*)(pln_b + n * D_DIM + c);
  const float4 pw = *(const float4*)(proto_w + row * D_DIM + c);
  float4 o;
  o.x = (v.x - mean) * inv * gg.x + bb.x + pw.x;
  o.y = (v.y - mean) * inv * gg.y + bb.y + pw.y;
  o.z = (v.z - mean) * inv * gg.z + bb.z + pw.z;
  o.w = (v.w - mean) * inv * gg.w + bb.w + pw.w;
  *(float4*)(prc + row * D_DIM + c) = o;
}

// causal softmax over one score row (2048 cols), in place
__global__ void __launch_bounds__(256) k_softmax(float* __restrict__ scores) {
  __shared__ float sb[256];
  float* sr = scores + (size_t)blockIdx.x * S_DIM;
  const int c = threadIdx.x * 8;
  float4 a = *(const float4*)(sr + c);
  float4 b = *(const float4*)(sr + c + 4);
  float m = fmaxf(fmaxf(fmaxf(a.x, a.y), fmaxf(a.z, a.w)),
                  fmaxf(fmaxf(b.x, b.y), fmaxf(b.z, b.w)));
  m = block_max(m, sb);
  a.x = __expf(a.x - m); a.y = __expf(a.y - m);
  a.z = __expf(a.z - m); a.w = __expf(a.w - m);
  b.x = __expf(b.x - m); b.y = __expf(b.y - m);
  b.z = __expf(b.z - m); b.w = __expf(b.w - m);
  float s = a.x + a.y + a.z + a.w + b.x + b.y + b.z + b.w;
  s = block_sum(s, sb);
  const float inv = 1.0f / s;
  a.x *= inv; a.y *= inv; a.z *= inv; a.w *= inv;
  b.x *= inv; b.y *= inv; b.z *= inv; b.w *= inv;
  *(float4*)(sr + c) = a;
  *(float4*)(sr + c + 4) = b;
}

// ---------------------------------------------------------------------------
// WMMA GEMM kernels
// ---------------------------------------------------------------------------
// prc[n] = in_proto[n] (D x D) . pt_w[n]^T        grid: 4 * 8 * 16 = 512
__global__ void __launch_bounds__(256) k_proto_gemm(const float* __restrict__ in_proto,
                                                    const float* __restrict__ pt_w,
                                                    float* __restrict__ prc) {
  const int blk = blockIdx.x;
  const int n = blk >> 7, t = blk & 127, mt = t >> 4, nt = t & 15;
  const float* A = in_proto + (size_t)n * D_DIM * D_DIM;
  const float* B = pt_w + (size_t)n * D_DIM * D_DIM;
  float* out = prc + (size_t)n * D_DIM * D_DIM;
  gemm_nt<false, false>(A, D_DIM, B, nullptr, D_DIM, D_DIM, mt * BM, nt * BN,
    [&](int r, int c, float a0, float) { out[(size_t)r * D_DIM + c] = a0; });
}

// masked[n] = silu(h.mu_w[n]^T + mu_b[n]) * ((h.proto_state[n]^T*scale - cost[n]) > 0)
// grid: 3 * 64 * 16 = 3072
__global__ void __launch_bounds__(256) k_expert_gemm(const float* __restrict__ h,
                                                     const float* __restrict__ mu_w,
                                                     const float* __restrict__ mu_b,
                                                     const float* __restrict__ proto_state,
                                                     const float* __restrict__ cost,
                                                     float* __restrict__ qkv) {
  const int blk = blockIdx.x;
  const int n = blk / 1024, t = blk % 1024, mt = t >> 4, nt = t & 15;
  const float* B0 = mu_w + (size_t)n * D_DIM * D_DIM;
  const float* B1 = proto_state + (size_t)n * D_DIM * D_DIM;
  const float* bias = mu_b + n * D_DIM;
  const float* cst  = cost + n * D_DIM;
  float* out = qkv + (size_t)n * M_TOK * D_DIM;
  gemm_nt<true, false>(h, D_DIM, B0, B1, D_DIM, D_DIM, mt * BM, nt * BN,
    [&](int r, int c, float a0, float a1) {
      const float s  = silu_f(a0 + bias[c]);
      const float z2 = a1 * SCALE_R - cst[c];
      out[(size_t)r * D_DIM + c] = (z2 > 0.0f) ? s : 0.0f;
    });
}

// scores[b] = q[b].k[b]^T * scale, causal-masked     grid: 4 * 16 * 32 = 2048
__global__ void __launch_bounds__(256) k_scores(const float* __restrict__ qkv,
                                                float* __restrict__ scores) {
  const int blk = blockIdx.x;
  const int b = blk >> 9, t = blk & 511, mt = t >> 5, nt = t & 31;
  const float* q = qkv + (size_t)b * S_DIM * D_DIM;
  const float* k = qkv + (size_t)M_TOK * D_DIM + (size_t)b * S_DIM * D_DIM;
  float* out = scores + (size_t)b * S_DIM * S_DIM;
  gemm_nt<false, false>(q, D_DIM, k, nullptr, D_DIM, D_DIM, mt * BM, nt * BN,
    [&](int r, int c, float a0, float) {
      out[(size_t)r * S_DIM + c] = (c > r) ? -3.0e38f : a0 * SCALE_R;
    });
}

// attn_out[b] = attn[b] . v[b]  (v is KxN -> transposed staging)
// grid: 4 * 16 * 16 = 1024
__global__ void __launch_bounds__(256) k_attn_v(const float* __restrict__ attn,
                                                const float* __restrict__ qkv,
                                                float* __restrict__ attn_out) {
  const int blk = blockIdx.x;
  const int b = blk >> 8, t = blk & 255, mt = t >> 4, nt = t & 15;
  const float* A = attn + (size_t)b * S_DIM * S_DIM;
  const float* V = qkv + 2ull * M_TOK * D_DIM + (size_t)b * S_DIM * D_DIM;
  float* out = attn_out + (size_t)b * S_DIM * D_DIM;
  gemm_nt<false, true>(A, S_DIM, V, nullptr, D_DIM, S_DIM, mt * BM, nt * BN,
    [&](int r, int c, float a0, float) { out[(size_t)r * D_DIM + c] = a0; });
}

// out = x + silu(ao.mu_w[3]^T + mu_b[3]) * ((ao.proto_state[3]^T*scale - cost[3]) > 0)
// grid: 64 * 16 = 1024
__global__ void __launch_bounds__(256) k_out_gemm(const float* __restrict__ attn_out,
                                                  const float* __restrict__ x,
                                                  const float* __restrict__ mu_w,
                                                  const float* __restrict__ mu_b,
                                                  const float* __restrict__ proto_state,
                                                  const float* __restrict__ cost,
                                                  float* __restrict__ out) {
  const int blk = blockIdx.x;
  const int mt = blk >> 4, nt = blk & 15;
  const float* B0 = mu_w + 3ull * D_DIM * D_DIM;
  const float* B1 = proto_state + 3ull * D_DIM * D_DIM;
  const float* bias = mu_b + 3 * D_DIM;
  const float* cst  = cost + 3 * D_DIM;
  gemm_nt<true, false>(attn_out, D_DIM, B0, B1, D_DIM, D_DIM, mt * BM, nt * BN,
    [&](int r, int c, float a0, float a1) {
      const float s  = silu_f(a0 + bias[c]);
      const float z2 = a1 * SCALE_R - cst[c];
      const size_t idx = (size_t)r * D_DIM + c;
      out[idx] = x[idx] + ((z2 > 0.0f) ? s : 0.0f);
    });
}

// ---------------------------------------------------------------------------
// Launch. Workspace (f32): cost 4K | h 8M | proto 4M | qkv 24M | scores 16M |
// attn_out 8M  ->  ~240 MB total. All buffers fully written before being read.
// ---------------------------------------------------------------------------
extern "C" void kernel_launch(void* const* d_in, const int* in_sizes, int n_in,
                              void* d_out, int out_size, void* d_ws, size_t ws_size,
                              hipStream_t stream) {
  const float* x        = (const float*)d_in[0];
  const float* ln1_g    = (const float*)d_in[1];
  const float* ln1_b    = (const float*)d_in[2];
  const float* mu_w     = (const float*)d_in[3];
  const float* mu_b     = (const float*)d_in[4];
  const float* proto_w  = (const float*)d_in[5];
  const float* gate     = (const float*)d_in[6];
  const float* pt_w     = (const float*)d_in[7];
  const float* pln_g    = (const float*)d_in[8];
  const float* pln_b    = (const float*)d_in[9];
  const float* in_proto = (const float*)d_in[10];
  float* out = (float*)d_out;

  float* ws      = (float*)d_ws;
  float* cost    = ws;                                       // 4*1024
  float* h       = cost + 4 * D_DIM;                         // 8M
  float* proto   = h + (size_t)M_TOK * D_DIM;                // 4M (prc -> proto_state)
  float* qkv     = proto + 4ull * D_DIM * D_DIM;             // 24M (q,k,v)
  float* scores  = qkv + 3ull * M_TOK * D_DIM;               // 16M (softmax in place)
  float* attn_o  = scores + (size_t)B_DIM * S_DIM * S_DIM;   // 8M

  k_cost      <<<4,              256, 0, stream>>>(gate, cost);
  k_ln_x      <<<M_TOK,          256, 0, stream>>>(x, ln1_g, ln1_b, h);
  k_proto_gemm<<<512,            256, 0, stream>>>(in_proto, pt_w, proto);
  k_proto_ln  <<<4 * D_DIM,      256, 0, stream>>>(proto, proto_w, pln_g, pln_b);
  k_expert_gemm<<<3072,          256, 0, stream>>>(h, mu_w, mu_b, proto, cost, qkv);
  k_scores    <<<2048,           256, 0, stream>>>(qkv, scores);
  k_softmax   <<<B_DIM * S_DIM,  256, 0, stream>>>(scores);
  k_attn_v    <<<1024,           256, 0, stream>>>(scores, qkv, attn_o);
  k_out_gemm  <<<1024,           256, 0, stream>>>(attn_o, x, mu_w, mu_b, proto, cost, out);
}
